// MultiHeadAttention_9466107921061
// MI455X (gfx1250) — matrix operand order repaired and checked by reading, not compiled, not run on hardware
//
#include <hip/hip_runtime.h>
#include <hip/hip_bf16.h>
#include <cstdint>

// ---------------- problem constants ----------------
constexpr int S_   = 2048;
constexpr int E_   = 2048;
constexpr int H_   = 16;
constexpr int KVH_ = 4;
constexpr int D_   = 128;
constexpr int F_   = H_ * D_ + 2 * KVH_ * D_;   // 3072 fused qkv width
constexpr float SCALE_ = 0.088388347648318447f; // 1/sqrt(128)

typedef unsigned short ushort_t;

// ---------------- WMMA types ----------------
typedef __attribute__((ext_vector_type(16))) __bf16 v16bf;
typedef __attribute__((ext_vector_type(8)))  float  v8f;
typedef __attribute__((ext_vector_type(4)))  int    v4i_t;

union Frag16 { uint4 q[2]; v16bf v; };

__device__ inline v8f zero8() { v8f z; for (int i = 0; i < 8; ++i) z[i] = 0.0f; return z; }

__device__ inline uint32_t f2bf(float f) {
    uint32_t u = __float_as_uint(f);
    return (u + 0x7FFFu + ((u >> 16) & 1u)) >> 16;   // RNE
}
__device__ inline uint32_t pk(float a, float b) { return f2bf(a) | (f2bf(b) << 16); }

__device__ inline v8f wmma_bf16(v16bf a, v16bf b, v8f c) {
    return __builtin_amdgcn_wmma_f32_16x16x32_bf16(false, a, false, b, (short)0, c, false, false);
}

// ---------------- async copy to LDS (CDNA5 path, guarded) ----------------
#if __has_builtin(__builtin_amdgcn_global_load_async_to_lds_b128) && \
    __has_builtin(__builtin_amdgcn_s_wait_asynccnt)
#define HAVE_ASYNC_LDS 1
#else
#define HAVE_ASYNC_LDS 0
#endif

typedef v4i_t __attribute__((address_space(1)))* gv4i_p;   // global int4*
typedef v4i_t __attribute__((address_space(3)))* lv4i_p;   // LDS int4*

__device__ inline void cp16_g2l(void* lds, const void* g) {
#if HAVE_ASYNC_LDS
    __builtin_amdgcn_global_load_async_to_lds_b128((gv4i_p)g, (lv4i_p)lds, 0, 0);
#else
    *(uint4*)lds = *(const uint4*)g;
#endif
}
__device__ inline void async_fence() {
#if HAVE_ASYNC_LDS
    __builtin_amdgcn_s_wait_asynccnt(0);
#endif
}

// A-fragment (16x32 bf16): lane<16 -> row m, K 0..15; lane>=16 -> K 16..31
__device__ inline v16bf lds_afrag(const uint32_t* tile /*stride 16 dwords/row*/, int m, int hi) {
    const uint32_t* p = tile + m * 16 + hi * 8;
    Frag16 f; f.q[0] = *(const uint4*)p; f.q[1] = *(const uint4*)(p + 4);
    return f.v;
}
// B-fragment (32x16 bf16): lane n holds source row n, K 0..15 (lo) / 16..31 (hi)
__device__ inline v16bf lds_bfrag(const uint32_t* rowp) {
    Frag16 f; f.q[0] = *(const uint4*)rowp; f.q[1] = *(const uint4*)(rowp + 4);
    return f.v;
}

// =====================================================================
// f32 -> bf16 bulk convert (vectorized, memory bound)
// =====================================================================
__global__ __launch_bounds__(256) void cvt_bf16(const float* __restrict__ src,
                                                ushort_t* __restrict__ dst) {
    int i = blockIdx.x * blockDim.x + threadIdx.x;   // one float4 per thread
    float4 f = ((const float4*)src)[i];
    uint2 o; o.x = pk(f.x, f.y); o.y = pk(f.z, f.w);
    ((uint2*)dst)[i] = o;
}

// =====================================================================
// GEMM: C[M,N] = A[M,K] * B[N,K]^T, A/B bf16 row-major, C f32.
// 256 threads (8 waves), block tile 128x128, wave tile 32x64, k-step 32.
// Requires M%128==0, N%128==0, K%32==0.
// =====================================================================
__global__ __launch_bounds__(256) void gemm_bf16_tn(const ushort_t* __restrict__ A,
                                                    const ushort_t* __restrict__ B,
                                                    float* __restrict__ C,
                                                    int M, int N, int K) {
    __shared__ __align__(16) uint32_t sA[128 * 16]; // 128 rows x 32 bf16 (8 KB)
    __shared__ __align__(16) uint32_t sB[128 * 16];

    const int tid  = threadIdx.x;
    const int lane = tid & 31, wave = tid >> 5;
    const int hi   = lane >> 4, lm = lane & 15;
    const int bm = blockIdx.y * 128, bn = blockIdx.x * 128;
    const int wm = (wave >> 1) * 32, wn = (wave & 1) * 64;

    v8f acc[2][4];
    for (int i = 0; i < 2; ++i) for (int j = 0; j < 4; ++j) acc[i][j] = zero8();

    const int srow = tid >> 1;          // 0..127
    const int sseg = (tid & 1) << 4;    // 0 or 16 (elements)

    for (int k0 = 0; k0 < K; k0 += 32) {
        {   // stage A/B tiles: pure 16B copies (async-to-LDS when available)
            const ushort_t* pa = A + (size_t)(bm + srow) * K + k0 + sseg;
            uint32_t* da = &sA[srow * 16 + (sseg >> 1)];
            cp16_g2l(da, pa); cp16_g2l(da + 4, pa + 8);
            const ushort_t* pb = B + (size_t)(bn + srow) * K + k0 + sseg;
            uint32_t* db = &sB[srow * 16 + (sseg >> 1)];
            cp16_g2l(db, pb); cp16_g2l(db + 4, pb + 8);
        }
        async_fence();
        __syncthreads();

        v16bf a0 = lds_afrag(sA, wm + lm, hi);
        v16bf a1 = lds_afrag(sA, wm + 16 + lm, hi);
        v16bf b0 = lds_bfrag(&sB[(wn +  0 + lm) * 16 + hi * 8]);
        v16bf b1 = lds_bfrag(&sB[(wn + 16 + lm) * 16 + hi * 8]);
        v16bf b2 = lds_bfrag(&sB[(wn + 32 + lm) * 16 + hi * 8]);
        v16bf b3 = lds_bfrag(&sB[(wn + 48 + lm) * 16 + hi * 8]);

        acc[0][0] = wmma_bf16(a0, b0, acc[0][0]);
        acc[0][1] = wmma_bf16(a0, b1, acc[0][1]);
        acc[0][2] = wmma_bf16(a0, b2, acc[0][2]);
        acc[0][3] = wmma_bf16(a0, b3, acc[0][3]);
        acc[1][0] = wmma_bf16(a1, b0, acc[1][0]);
        acc[1][1] = wmma_bf16(a1, b1, acc[1][1]);
        acc[1][2] = wmma_bf16(a1, b2, acc[1][2]);
        acc[1][3] = wmma_bf16(a1, b3, acc[1][3]);
        __syncthreads();
    }
    for (int i = 0; i < 2; ++i)
        for (int j = 0; j < 4; ++j)
            for (int r = 0; r < 8; ++r)
                C[(size_t)(bm + wm + i * 16 + r + 8 * hi) * N + (bn + wn + j * 16 + lm)] =
                    acc[i][j][r];
}

// =====================================================================
// RoPE + bf16 emit: reads qkv f32, writes bf16 q/k (roped, fused layout)
// and bf16 V^T [kvh*128+d][s] for conflict-free flash staging.
// idx -> (s, hh, j): hh 0..15 q-heads, 16..19 k-heads, 20..23 v-heads.
// =====================================================================
__global__ __launch_bounds__(256) void rope_convert(const float* __restrict__ qkv,
                                                    ushort_t* __restrict__ qkvb,
                                                    ushort_t* __restrict__ vT) {
    int idx = blockIdx.x * blockDim.x + threadIdx.x;   // S * 24 * 64
    int j  = idx & 63;
    int hh = (idx >> 6) % 24;
    int s  = idx / (64 * 24);
    if (hh < 20) {
        int off = (hh < H_) ? hh * D_ : H_ * D_ + (hh - H_) * D_;
        const float* base = qkv + (size_t)s * F_ + off;
        float inv_freq = __expf(-((float)j / 64.0f) * 9.2103403719761836f); // ln(1e4)
        float sn, cs;
        __sincosf((float)s * inv_freq, &sn, &cs);
        float x1 = base[j], x2 = base[j + 64];
        ushort_t* ob = qkvb + (size_t)s * F_ + off;
        ob[j]      = (ushort_t)f2bf(x1 * cs - x2 * sn);
        ob[j + 64] = (ushort_t)f2bf(x2 * cs + x1 * sn);
    } else {
        int vh = hh - 20;
        const float* base = qkv + (size_t)s * F_ + (H_ + KVH_) * D_ + vh * D_;
        vT[(size_t)(vh * D_ + j)      * S_ + s] = (ushort_t)f2bf(base[j]);
        vT[(size_t)(vh * D_ + j + 64) * S_ + s] = (ushort_t)f2bf(base[j + 64]);
    }
}

// =====================================================================
// Flash attention (causal, GQA rep=4) on bf16 Q/K/V^T, f32 accumulate.
// Block: 128 threads (4 waves) = head h, query rows [q0, q0+64).
// =====================================================================
__global__ __launch_bounds__(128) void flash_attn_gqa(const ushort_t* __restrict__ qkvb,
                                                      const ushort_t* __restrict__ vT,
                                                      ushort_t* __restrict__ ctxb) {
    __shared__ __align__(16) uint32_t       sK[32 * 64];   // 32 keys x 128 bf16 (8 KB)
    __shared__ __align__(16) uint32_t       sV[128 * 16];  // V^T: 128 d x 32 keys (8 KB)
    __shared__ __align__(16) unsigned short sP[4][16 * 32];// per-wave P relayout scratch

    const int h  = blockIdx.y;
    const int q0 = blockIdx.x * 64;
    const int kh = h >> 2;
    const int tid = threadIdx.x, lane = tid & 31, wave = tid >> 5;
    const int hi = lane >> 4, lm = lane & 15;
    const int qrow_base = q0 + wave * 16;

    // ---- Q tile -> A-fragments (pure b128 loads of bf16) ----
    v16bf qf[4];
    {
        const ushort_t* qsrc = qkvb + (size_t)(qrow_base + lm) * F_ + h * D_ + hi * 16;
        for (int c = 0; c < 4; ++c) {
            Frag16 fr;
            fr.q[0] = *(const uint4*)(qsrc + c * 32);
            fr.q[1] = *(const uint4*)(qsrc + c * 32 + 8);
            qf[c] = fr.v;
        }
    }

    v8f o[8];
    for (int cb = 0; cb < 8; ++cb) o[cb] = zero8();
    float mi[8], li[8];
    for (int r = 0; r < 8; ++r) { mi[r] = -__builtin_inff(); li[r] = 0.0f; }

    const int kend = q0 + 64;
    const int skey = tid >> 2;           // 0..31 (K staging row)
    const int sseg = (tid & 3) << 5;     // 0,32,64,96 (elements)

    for (int kb = 0; kb < kend; kb += 32) {
        {   // stage K rows + V^T rows: contiguous 64B per thread, async path
            const ushort_t* ks = qkvb + (size_t)(kb + skey) * F_ + H_ * D_ + kh * D_ + sseg;
            uint32_t* dk = &sK[skey * 64 + (sseg >> 1)];
            cp16_g2l(dk,      ks);      cp16_g2l(dk + 4,  ks + 8);
            cp16_g2l(dk + 8,  ks + 16); cp16_g2l(dk + 12, ks + 24);
            const ushort_t* vs = vT + (size_t)(kh * D_ + tid) * S_ + kb;
            uint32_t* dv = &sV[tid * 16];
            cp16_g2l(dv,      vs);      cp16_g2l(dv + 4,  vs + 8);
            cp16_g2l(dv + 8,  vs + 16); cp16_g2l(dv + 12, vs + 24);
        }
        async_fence();
        __syncthreads();

        // ---- S = Q K^T : two 16x16 tiles ----
        v8f s0 = zero8(), s1 = zero8();
        for (int c = 0; c < 4; ++c) {
            v16bf b0 = lds_bfrag(&sK[(lm)      * 64 + c * 16 + hi * 8]);
            v16bf b1 = lds_bfrag(&sK[(16 + lm) * 64 + c * 16 + hi * 8]);
            s0 = wmma_bf16(qf[c], b0, s0);
            s1 = wmma_bf16(qf[c], b1, s1);
        }

        // ---- online softmax (row = r + 8*hi, col = lm) ----
        float mnew[8], rsum[8];
        for (int r = 0; r < 8; ++r) {
            const int row = qrow_base + 8 * hi + r;
            float a = (kb + lm      <= row) ? s0[r] * SCALE_ : -__builtin_inff();
            float b = (kb + 16 + lm <= row) ? s1[r] * SCALE_ : -__builtin_inff();
            float m = fmaxf(a, b);
            for (int off = 1; off < 16; off <<= 1) m = fmaxf(m, __shfl_xor(m, off, 32));
            m = fmaxf(m, mi[r]);
            mnew[r] = m;
            float pa = __expf(a - m), pb = __expf(b - m);
            s0[r] = pa; s1[r] = pb;
            float rs = pa + pb;
            for (int off = 1; off < 16; off <<= 1) rs += __shfl_xor(rs, off, 32);
            rsum[r] = rs;
        }
        for (int r = 0; r < 8; ++r) {
            float alpha = __expf(mi[r] - mnew[r]);
            li[r] = li[r] * alpha + rsum[r];
            mi[r] = mnew[r];
            for (int cb = 0; cb < 8; ++cb) o[cb][r] *= alpha;
        }

        // ---- P: C layout -> A layout via per-wave LDS scratch ----
        {
            unsigned short* pw = &sP[wave][0];
            for (int r = 0; r < 8; ++r) {
                const int row = r + 8 * hi;
                pw[row * 32 + lm]      = (unsigned short)f2bf(s0[r]);
                pw[row * 32 + 16 + lm] = (unsigned short)f2bf(s1[r]);
            }
        }
        __syncthreads();   // uniform across block (same kb count for all waves)

        v16bf pf = lds_afrag((const uint32_t*)&sP[wave][0], lm, hi);

        // ---- O += P V ----
        for (int cb = 0; cb < 8; ++cb) {
            v16bf vf = lds_bfrag(&sV[(cb * 16 + lm) * 16 + hi * 8]);
            o[cb] = wmma_bf16(pf, vf, o[cb]);
        }
        __syncthreads();
    }

    // ---- finalize: O /= l, emit bf16 ctx[s, h*D + d] ----
    for (int r = 0; r < 8; ++r) {
        float inv = 1.0f / li[r];
        for (int cb = 0; cb < 8; ++cb) o[cb][r] *= inv;
    }
    for (int cb = 0; cb < 8; ++cb)
        for (int r = 0; r < 8; ++r)
            ctxb[(size_t)(qrow_base + r + 8 * hi) * (H_ * D_) + h * D_ + cb * 16 + lm] =
                (ushort_t)f2bf(o[cb][r]);
}

// =====================================================================
// launch
// =====================================================================
extern "C" void kernel_launch(void* const* d_in, const int* in_sizes, int n_in,
                              void* d_out, int out_size, void* d_ws, size_t ws_size,
                              hipStream_t stream) {
    const float* x       = (const float*)d_in[0];   // [S, E]
    const float* w_qkv   = (const float*)d_in[1];   // [F, E]
    const float* w_dense = (const float*)d_in[2];   // [E, H*D]
    float* out = (float*)d_out;                     // [S, E]

    char* ws = (char*)d_ws;
    size_t off = 0;
    ushort_t* x_bf    = (ushort_t*)(ws + off); off += (size_t)S_ * E_ * 2;        // 8 MB
    ushort_t* wqkv_bf = (ushort_t*)(ws + off); off += (size_t)F_ * E_ * 2;        // 12 MB
    ushort_t* wd_bf   = (ushort_t*)(ws + off); off += (size_t)E_ * H_ * D_ * 2;   // 8 MB
    float*    qkv     = (float*)   (ws + off); off += (size_t)S_ * F_ * 4;        // 25 MB
    ushort_t* qkv_bf  = (ushort_t*)(ws + off); off += (size_t)S_ * F_ * 2;        // 12.6 MB
    ushort_t* vT      = (ushort_t*)(ws + off); off += (size_t)KVH_ * D_ * S_ * 2; // 2 MB
    ushort_t* ctx_bf  = (ushort_t*)(ws + off);                                    // 8 MB

    // 0) one-shot f32 -> bf16 conversions (memory-bound pre-pass)
    cvt_bf16<<<(S_ * E_ / 4) / 256, 256, 0, stream>>>(x, x_bf);
    cvt_bf16<<<(F_ * E_ / 4) / 256, 256, 0, stream>>>(w_qkv, wqkv_bf);
    cvt_bf16<<<(E_ * H_ * D_ / 4) / 256, 256, 0, stream>>>(w_dense, wd_bf);

    // 1) qkv = x @ w_qkv^T   (f32 out for RoPE precision)
    {
        dim3 grid(F_ / 128, S_ / 128);
        gemm_bf16_tn<<<grid, 256, 0, stream>>>(x_bf, wqkv_bf, qkv, S_, F_, E_);
    }
    // 2) RoPE q/k -> bf16 fused buffer; V -> bf16 transposed buffer
    {
        int total = S_ * 24 * 64;
        rope_convert<<<total / 256, 256, 0, stream>>>(qkv, qkv_bf, vT);
    }
    // 3) flash attention -> bf16 ctx
    {
        dim3 grid(S_ / 64, H_);
        flash_attn_gqa<<<grid, 128, 0, stream>>>(qkv_bf, vT, ctx_bf);
    }
    // 4) out = ctx @ w_dense^T
    {
        dim3 grid(E_ / 128, S_ / 128);
        gemm_bf16_tn<<<grid, 256, 0, stream>>>(ctx_bf, wd_bf, out, S_, E_, H_ * D_);
    }
}
